// Attention_5823975653733
// MI455X (gfx1250) — compile-verified, matrix operand order checked by reading, not run on hardware
//
#include <hip/hip_runtime.h>

// ---------------------------------------------------------------------------
// MI455X (gfx1250) fused mixer-block:
//   K0: fc2_w / proj_w  f32 -> bf16 into workspace (L2-resident bf16 streams)
//   K1: one workgroup per batch b (512 WGs, 512 thr = 16 waves):
//         x -> LDS bf16, per-channel 8x8 LN/fc1/gelu mixing (x_h, x_w),
//         LN2 in-place (wave32 shuffle reduce), WMMA bf16 GEMM1 (64x2304x768)
//         with batched B-frag loads (12 loads -> 12 back-to-back WMMAs),
//         gelu, Y bf16 -> workspace.  x_fuse lives entirely in 297KB LDS.
//   K2: WMMA bf16 GEMM2 (32768x768x768) + bias; B tile staged into LDS with
//       CDNA5 async global->LDS copies (ASYNCcnt + s_wait_asynccnt).
// ---------------------------------------------------------------------------

#define B_   512
#define N_   64
#define C_   768
#define C3_  2304
#define LDST 2320   // padded LDS row stride (bf16 elems): 1160 dwords % 64 banks = 8

typedef __attribute__((ext_vector_type(16))) __bf16 v16bf;
typedef __attribute__((ext_vector_type(8)))  float  v8f;

struct __align__(16) Q16 { unsigned int w[4]; };
union AFrag { Q16 q[2]; v16bf v; };
union F8    { v8f v; float f[8]; };

__device__ __forceinline__ float gelu_tanh(float x) {
    return 0.5f * x * (1.0f + tanhf(0.79788456080286535588f * (x + 0.044715f * x * x * x)));
}

// CDNA5 async global->LDS copy (tracked by ASYNCcnt), via inline asm so it is
// portable across ROCm-7.2 / amdgpu-toolchain builtin arity differences.
__device__ __forceinline__ void async_g2l_b128(const __bf16* gsrc, __bf16* ldst) {
    unsigned lds_lo = (unsigned)(uintptr_t)ldst;   // generic LDS addr[31:0] = LDS offset
    asm volatile("global_load_async_to_lds_b128 %0, %1, off"
                 :: "v"(lds_lo), "v"(gsrc) : "memory");
}
__device__ __forceinline__ void wait_async0() {
    asm volatile("s_wait_asynccnt 0" ::: "memory");
}

// --------------------------- K0: weight convert ----------------------------
__global__ void cvt_w_bf16(const float* __restrict__ fc2w, const float* __restrict__ projw,
                           __bf16* __restrict__ fc2w_bf, __bf16* __restrict__ projw_bf) {
    int i = blockIdx.x * blockDim.x + threadIdx.x;
    const int n1 = C_ * C3_;   // 1,769,472
    const int n2 = C_ * C_;    //   589,824
    if (i < n1) fc2w_bf[i]  = (__bf16)fc2w[i];
    if (i < n2) projw_bf[i] = (__bf16)projw[i];
}

// --------------------------- K1: fused per-batch ---------------------------
__global__ __launch_bounds__(512, 1)
void fused_mixer(const float* __restrict__ xg,
                 const float* __restrict__ ln1w, const float* __restrict__ ln1b,
                 const float* __restrict__ fc1w,
                 const float* __restrict__ ln2w, const float* __restrict__ ln2b,
                 const __bf16* __restrict__ fc2w_bf,
                 __bf16* __restrict__ Y) {
    __shared__ __align__(16) __bf16 Xf[N_][LDST];   // 64 x 2320 bf16 = 296,960 B

    const int tid = threadIdx.x;
    const int b   = blockIdx.x;
    const float* xb = xg + (size_t)b * N_ * C_;

    // ---- Phase 0: x[b] -> LDS bf16 (features 0..767 of x_fuse) ----
    for (int idx = tid; idx < N_ * C_; idx += 512) {
        int t = idx / C_, c = idx % C_;
        Xf[t][c] = (__bf16)xb[idx];
    }
    __syncthreads();

    // ---- Phase A: per-channel 8x8 token mixing -> features 768..2303 ----
    for (int c = tid; c < C_; c += 512) {
        float X[8][8];
        #pragma unroll
        for (int i = 0; i < 8; ++i)
            #pragma unroll
            for (int j = 0; j < 8; ++j)
                X[i][j] = (float)Xf[i * 8 + j][c];

        // Fw = gelu(LN_rows(X) @ fc1^T);  x_w = Fw @ X
        float Fw[8][8];
        #pragma unroll
        for (int i = 0; i < 8; ++i) {
            float m = 0.f;
            #pragma unroll
            for (int j = 0; j < 8; ++j) m += X[i][j];
            m *= 0.125f;
            float v = 0.f;
            #pragma unroll
            for (int j = 0; j < 8; ++j) { float d = X[i][j] - m; v += d * d; }
            float rs = rsqrtf(v * 0.125f + 1e-5f);
            float nrm[8];
            #pragma unroll
            for (int j = 0; j < 8; ++j) nrm[j] = (X[i][j] - m) * rs * ln1w[j] + ln1b[j];
            #pragma unroll
            for (int j2 = 0; j2 < 8; ++j2) {
                float a = 0.f;
                #pragma unroll
                for (int k = 0; k < 8; ++k) a += nrm[k] * fc1w[j2 * 8 + k];
                Fw[i][j2] = gelu_tanh(a);
            }
        }
        #pragma unroll
        for (int i = 0; i < 8; ++i)
            #pragma unroll
            for (int j = 0; j < 8; ++j) {
                float s = 0.f;
                #pragma unroll
                for (int k = 0; k < 8; ++k) s += Fw[i][k] * X[k][j];
                Xf[i * 8 + j][2 * C_ + c] = (__bf16)s;   // x_w
            }

        // x_h = X @ Fh, with Fh[k][j] = G[j][k], G = gelu(LN(X^T) @ fc1^T)
        #pragma unroll
        for (int j = 0; j < 8; ++j) {
            float m = 0.f;
            #pragma unroll
            for (int i = 0; i < 8; ++i) m += X[i][j];
            m *= 0.125f;
            float v = 0.f;
            #pragma unroll
            for (int i = 0; i < 8; ++i) { float d = X[i][j] - m; v += d * d; }
            float rs = rsqrtf(v * 0.125f + 1e-5f);
            float nrm[8];
            #pragma unroll
            for (int i = 0; i < 8; ++i) nrm[i] = (X[i][j] - m) * rs * ln1w[i] + ln1b[i];
            float g[8];
            #pragma unroll
            for (int k = 0; k < 8; ++k) {
                float a = 0.f;
                #pragma unroll
                for (int i = 0; i < 8; ++i) a += nrm[i] * fc1w[k * 8 + i];
                g[k] = gelu_tanh(a);
            }
            #pragma unroll
            for (int i = 0; i < 8; ++i) {
                float s = 0.f;
                #pragma unroll
                for (int k = 0; k < 8; ++k) s += X[i][k] * g[k];
                Xf[i * 8 + j][C_ + c] = (__bf16)s;       // x_h
            }
        }
    }
    __syncthreads();

    // ---- Phase B: LayerNorm(2304) in place, one wave32 per row ----
    const int lane = tid & 31, wave = tid >> 5;
    for (int row = wave; row < N_; row += 16) {
        float s = 0.f, s2 = 0.f;
        for (int k = lane; k < C3_; k += 32) {
            float v = (float)Xf[row][k];
            s += v; s2 += v * v;
        }
        #pragma unroll
        for (int off = 16; off > 0; off >>= 1) {
            s  += __shfl_xor(s,  off, 32);
            s2 += __shfl_xor(s2, off, 32);
        }
        float m  = s * (1.f / C3_);
        float va = s2 * (1.f / C3_) - m * m;
        float rs = rsqrtf(va + 1e-5f);
        for (int k = lane; k < C3_; k += 32) {
            float v = (float)Xf[row][k];
            Xf[row][k] = (__bf16)((v - m) * rs * ln2w[k] + ln2b[k]);
        }
    }
    __syncthreads();

    // ---- Phase C: WMMA GEMM1  [64 x 2304] @ fc2_w^T -> gelu -> Y bf16 ----
    // 16 waves: wave -> (m-tile = wave&3, n-block = wave>>2 of 12 N-tiles)
    const int m  = wave & 3;
    const int nb = wave >> 2;
    const int lh = lane & 15;
    const int kh = (lane < 16) ? 0 : 8;

    F8 acc[12];
    #pragma unroll
    for (int t = 0; t < 12; ++t)
        #pragma unroll
        for (int r = 0; r < 8; ++r) acc[t].f[r] = 0.f;

    const __bf16* Arow = &Xf[m * 16 + lh][0];
    const __bf16* Bp[12];
    #pragma unroll
    for (int t = 0; t < 12; ++t)
        Bp[t] = fc2w_bf + (size_t)((nb * 12 + t) * 16 + lh) * C3_ + kh;

    for (int k = 0; k < C3_; k += 32) {
        AFrag a;
        a.q[0] = *(const Q16*)(Arow + k + kh);          // K in {kh..kh+7}
        a.q[1] = *(const Q16*)(Arow + k + 16 + kh);     // K in {16+kh..}
        AFrag bb[12];
        #pragma unroll
        for (int t = 0; t < 12; ++t) {                  // batched loads -> 1 wait
            bb[t].q[0] = *(const Q16*)(Bp[t] + k);
            bb[t].q[1] = *(const Q16*)(Bp[t] + k + 16);
            __builtin_prefetch(Bp[t] + k + 256, 0, 3);
        }
        #pragma unroll
        for (int t = 0; t < 12; ++t)                    // 12 independent WMMAs
            acc[t].v = __builtin_amdgcn_wmma_f32_16x16x32_bf16(
                false, a.v, false, bb[t].v, (short)0, acc[t].v, false, false);
    }

    __bf16* Yb = Y + (size_t)b * N_ * C_;
    #pragma unroll
    for (int t = 0; t < 12; ++t) {
        const int col = (nb * 12 + t) * 16 + lh;
        #pragma unroll
        for (int r = 0; r < 8; ++r) {
            const int row = m * 16 + r + ((lane < 16) ? 0 : 8);
            Yb[row * C_ + col] = (__bf16)gelu_tanh(acc[t].f[r]);
        }
    }
}

// --------------------------- K2: projection GEMM ---------------------------
// out[32768 x 768] = Y @ proj_w^T + proj_b ; 256 thr = 8 waves, 128x128 tile/WG.
// B tile (128 rows x 32 k, bf16 = 8KB) staged via async global->LDS per k-step.
__global__ __launch_bounds__(256, 2)
void proj_gemm(const __bf16* __restrict__ Y, const __bf16* __restrict__ pw,
               const float* __restrict__ pb, float* __restrict__ out) {
    __shared__ __align__(16) __bf16 Bs[128][40];   // 10,240 B (pad: 80B row stride)

    const int tid  = threadIdx.x;
    const int lane = tid & 31, wave = tid >> 5;
    const int mBase = blockIdx.y * 128;
    const int nBase = blockIdx.x * 128;
    const int lh = lane & 15;
    const int kh = (lane < 16) ? 0 : 8;

    // this thread's two 16B staging chunks: chunk c -> row c>>2, col (c&3)*8
    const int c0 = tid, c1 = tid + 256;
    const int r0 = c0 >> 2, h0 = (c0 & 3) * 8;
    const int r1 = c1 >> 2, h1 = (c1 & 3) * 8;
    const __bf16* g0 = pw + (size_t)(nBase + r0) * C_ + h0;
    const __bf16* g1 = pw + (size_t)(nBase + r1) * C_ + h1;

    F8 acc[8];
    #pragma unroll
    for (int t = 0; t < 8; ++t)
        #pragma unroll
        for (int r = 0; r < 8; ++r) acc[t].f[r] = 0.f;

    const __bf16* Arow = Y + (size_t)(mBase + wave * 16 + lh) * C_;

    for (int k = 0; k < C_; k += 32) {
        __syncthreads();                       // previous iter's Bs reads done
        async_g2l_b128(g0 + k, &Bs[r0][h0]);   // ASYNCcnt-tracked global->LDS
        async_g2l_b128(g1 + k, &Bs[r1][h1]);
        __builtin_prefetch(g0 + k + 64, 0, 3); // warm next tile in WGP$/L2
        wait_async0();                         // s_wait_asynccnt 0
        __syncthreads();                       // tile visible to all 8 waves

        AFrag a;
        a.q[0] = *(const Q16*)(Arow + k + kh);
        a.q[1] = *(const Q16*)(Arow + k + 16 + kh);
        AFrag bb[8];
        #pragma unroll
        for (int t = 0; t < 8; ++t) {          // batched ds_read_b128 pairs
            bb[t].q[0] = *(const Q16*)(&Bs[t * 16 + lh][kh]);
            bb[t].q[1] = *(const Q16*)(&Bs[t * 16 + lh][16 + kh]);
        }
        #pragma unroll
        for (int t = 0; t < 8; ++t)
            acc[t].v = __builtin_amdgcn_wmma_f32_16x16x32_bf16(
                false, a.v, false, bb[t].v, (short)0, acc[t].v, false, false);
    }

    #pragma unroll
    for (int t = 0; t < 8; ++t) {
        const int col = nBase + t * 16 + lh;
        const float bias = pb[col];
        #pragma unroll
        for (int r = 0; r < 8; ++r) {
            const int row = mBase + wave * 16 + r + ((lane < 16) ? 0 : 8);
            out[(size_t)row * C_ + col] = acc[t].f[r] + bias;
        }
    }
}

// ------------------------------- launcher ----------------------------------
extern "C" void kernel_launch(void* const* d_in, const int* in_sizes, int n_in,
                              void* d_out, int out_size, void* d_ws, size_t ws_size,
                              hipStream_t stream) {
    const float* x     = (const float*)d_in[0];
    const float* ln1w  = (const float*)d_in[1];
    const float* ln1b  = (const float*)d_in[2];
    const float* fc1w  = (const float*)d_in[3];
    const float* ln2w  = (const float*)d_in[4];
    const float* ln2b  = (const float*)d_in[5];
    const float* fc2w  = (const float*)d_in[6];
    const float* projw = (const float*)d_in[7];
    const float* projb = (const float*)d_in[8];
    float* out = (float*)d_out;

    // workspace: [fc2_w bf16 | proj_w bf16 | Y bf16] = 3.4MB + 1.1MB + 48MB
    __bf16* fc2w_bf  = (__bf16*)d_ws;
    __bf16* projw_bf = fc2w_bf + (size_t)C_ * C3_;
    __bf16* Ybuf     = projw_bf + (size_t)C_ * C_;

    const int ncvt = C_ * C3_;
    cvt_w_bf16<<<(ncvt + 255) / 256, 256, 0, stream>>>(fc2w, projw, fc2w_bf, projw_bf);

    fused_mixer<<<B_, 512, 0, stream>>>(x, ln1w, ln1b, fc1w, ln2w, ln2b, fc2w_bf, Ybuf);

    dim3 g2(C_ / 128, (B_ * N_) / 128);   // (6, 256)
    proj_gemm<<<g2, 256, 0, stream>>>(Ybuf, projw_bf, projb, out);
}
